// VectorizedDINOv3Group_62431644615068
// MI455X (gfx1250) — matrix operand order repaired and checked by reading, not compiled, not run on hardware
//
#include <hip/hip_runtime.h>
#include <hip/hip_bf16.h>
#include <cstdint>

// Problem constants (from reference)
#define E_ 2
#define B_ 2
#define T_ 197
#define D_ 1024
#define H_ 16
#define DH_ 64
#define L_ 24
#define F_ 4096
#define C_ 4
#define NPATCH 196   // 14*14
#define PK_ 768      // 3*16*16

typedef __attribute__((ext_vector_type(16))) __bf16 v16bf;
typedef __attribute__((ext_vector_type(8)))  float  v8f;

// scalar f32 -> bf16 (round-to-nearest-even), used outside hot loops
__device__ __forceinline__ unsigned short f2bf(float f) {
  unsigned u = __builtin_bit_cast(unsigned, f);
  u += 0x7FFFu + ((u >> 16) & 1u);
  return (unsigned short)(u >> 16);
}

// packed pair f32x2 -> bf16x2 in one dword (hot path: weight conversion)
__device__ __forceinline__ unsigned pkbf(float a, float b) {
#if __has_builtin(__builtin_amdgcn_cvt_pk_bf16_f32)
  auto r = __builtin_amdgcn_cvt_pk_bf16_f32(a, b);   // v_cvt_pk_bf16_f32
  return __builtin_bit_cast(unsigned, r);
#else
  // round-half-up then byte-pack with a single v_perm_b32
  unsigned ua = __builtin_bit_cast(unsigned, a) + 0x8000u;
  unsigned ub = __builtin_bit_cast(unsigned, b) + 0x8000u;
  // dst = { ub[31:16], ua[31:16] }  (src0 bytes are sel 4-7, src1 bytes 0-3)
  return __builtin_amdgcn_perm(ub, ua, 0x07060302u);
#endif
}

union Frag { uint4 q[2]; v16bf v; };

__device__ __forceinline__ v8f bwmma(const Frag& a, const Frag& b, v8f c) {
  return __builtin_amdgcn_wmma_f32_16x16x32_bf16(false, a.v, false, b.v,
                                                 (short)0, c, false, false);
}

enum { EP_STORE = 0, EP_RESID = 1, EP_GELU = 2, EP_PATCH = 3 };

// C[M,N] = A[M,K](bf16) * W[N,K]^T(f32 -> bf16 on the fly) + bias, with epilogue.
// grid: (N/64, ceil(M/128), E). block: 256 = 8 waves, each wave a 32x32 output
// (4 x v_wmma_f32_16x16x32_bf16 per 32-K step). Double-buffered LDS, one
// barrier per K-step. A tile staged with global_load_async_to_lds_b128
// (ASYNCcnt); W tile staged through VGPRs (needs f32->bf16 conversion).
template<int EP>
__global__ __launch_bounds__(256)
void gemm_bf16_wmma(const unsigned short* __restrict__ A, size_t sA,
                    const float* __restrict__ W, size_t sW,
                    const float* __restrict__ bias, size_t sBias,
                    float* out, size_t sOut, int ldOut,
                    unsigned short* outbf,
                    const float* __restrict__ ls, size_t sLs,
                    const float* __restrict__ pos, size_t sPos,
                    int Mtot, int K)
{
  __shared__ __align__(16) unsigned short As[2][128 * 40]; // 128 M-rows x 32 K, pitch 40
  __shared__ __align__(16) unsigned short Ws[2][64 * 40];  // 64 N-rows x 32 K, pitch 40

  const int tid   = threadIdx.x;
  const int lane  = tid & 31;
  const int wv    = tid >> 5;
  const int waveM = wv & 3;        // 32-row subtile
  const int waveN = wv >> 2;       // 32-col subtile
  const int e  = blockIdx.z;
  const int m0 = blockIdx.y * 128;
  const int n0 = blockIdx.x * 64;

  const unsigned short* Ae = A + (size_t)e * sA;
  const float*          We = W + (size_t)e * sW;

  // A loader: thread -> (row = tid>>1, 32B chunk at (tid&1)*16 elements)
  const int arow  = tid >> 1;                 // 0..127
  const int ahalf = (tid & 1) * 16;           // element offset 0 / 16
  int am = m0 + arow; if (am >= Mtot) am = Mtot - 1;  // clamp; rows >= Mtot never stored
  const unsigned short* aSrc = Ae + (size_t)am * K + ahalf;
  // W loader: thread -> (row = tid>>2, 8 floats at (tid&3)*8)
  const int wrow = tid >> 2;                  // 0..63
  const int wseg = (tid & 3) * 8;
  const float* wSrc = We + (size_t)(n0 + wrow) * K + wseg;

  uint4 wb;
  auto loadW = [&](int k0) {
    const float4* wp = (const float4*)(wSrc + k0);
    float4 w0 = wp[0], w1 = wp[1];
    wb.x = pkbf(w0.x, w0.y); wb.y = pkbf(w0.z, w0.w);
    wb.z = pkbf(w1.x, w1.y); wb.w = pkbf(w1.z, w1.w);
  };
  auto storeW = [&](int buf) {
    *(uint4*)&Ws[buf][wrow * 40 + wseg] = wb;
  };
  auto asyncA = [&](int k0, int buf) {
    unsigned ldsA = (unsigned)(uintptr_t)&As[buf][arow * 40 + ahalf];
    const unsigned short* g = aSrc + k0;
    // copies 16B/lane straight into LDS; second chunk via shared inst offset
    asm volatile("global_load_async_to_lds_b128 %0, %1, off"
                 :: "v"(ldsA), "v"(g) : "memory");
    asm volatile("global_load_async_to_lds_b128 %0, %1, off offset:16"
                 :: "v"(ldsA), "v"(g) : "memory");
  };

  v8f acc00 = {}, acc01 = {}, acc10 = {}, acc11 = {};
  const int r  = lane & 15;
  const int hi = lane >> 4;

  // prologue: tile 0 into buffer 0
  asyncA(0, 0);
  loadW(0);
  storeW(0);

  int cur = 0;
  for (int k0 = 32;; k0 += 32) {
    const bool more = (k0 < K);
    if (more) loadW(k0);                       // global f32 loads + cvt overlap WMMA
    asm volatile("s_wait_asynccnt 0x0" ::: "memory");  // own async copies done
    __syncthreads();                           // everyone's tile visible

    const unsigned short* Ab = As[cur];
    const unsigned short* Wb = Ws[cur];
    Frag fa0, fa1, fb0, fb1;
    // A frag (16x32): lane half hi=0 -> K{0..7,16..23}, hi=1 -> K{8..15,24..31}
    fa0.q[0] = *(const uint4*)&Ab[(waveM * 32 + r) * 40 + hi * 8];
    fa0.q[1] = *(const uint4*)&Ab[(waveM * 32 + r) * 40 + 16 + hi * 8];
    fa1.q[0] = *(const uint4*)&Ab[(waveM * 32 + 16 + r) * 40 + hi * 8];
    fa1.q[1] = *(const uint4*)&Ab[(waveM * 32 + 16 + r) * 40 + 16 + hi * 8];
    // B frag (32x16): lane = column, hi=0 -> K{0..7,16..23}, hi=1 -> K{8..15,24..31}
    fb0.q[0] = *(const uint4*)&Wb[(waveN * 32 + r) * 40 + hi * 8];
    fb0.q[1] = *(const uint4*)&Wb[(waveN * 32 + r) * 40 + 16 + hi * 8];
    fb1.q[0] = *(const uint4*)&Wb[(waveN * 32 + 16 + r) * 40 + hi * 8];
    fb1.q[1] = *(const uint4*)&Wb[(waveN * 32 + 16 + r) * 40 + 16 + hi * 8];

    acc00 = bwmma(fa0, fb0, acc00);
    acc01 = bwmma(fa0, fb1, acc01);
    acc10 = bwmma(fa1, fb0, acc10);
    acc11 = bwmma(fa1, fb1, acc11);

    if (!more) break;
    // safe: previous readers of buffer cur^1 finished before the barrier above
    asyncA(k0, cur ^ 1);
    storeW(cur ^ 1);
    cur ^= 1;
  }

  // epilogue: VGPR i of each acc holds row (hi*8 + i), column = lane&15 (+16)
  const float* biasE = bias + (size_t)e * sBias;
  #pragma unroll
  for (int sub = 0; sub < 2; ++sub) {
    #pragma unroll
    for (int i = 0; i < 8; ++i) {
      int m = m0 + waveM * 32 + sub * 16 + hi * 8 + i;
      if (m >= Mtot) continue;
      #pragma unroll
      for (int half = 0; half < 2; ++half) {
        int col = n0 + waveN * 32 + half * 16 + r;
        float acc = sub == 0 ? (half ? acc01[i] : acc00[i])
                             : (half ? acc11[i] : acc10[i]);
        float v = acc + biasE[col];
        if constexpr (EP == EP_STORE) {
          out[(size_t)e * sOut + (size_t)m * ldOut + col] = v;
        } else if constexpr (EP == EP_RESID) {
          float* zp = out + (size_t)e * sOut + (size_t)m * ldOut + col;
          *zp += ls[(size_t)e * sLs + col] * v;
        } else if constexpr (EP == EP_GELU) {
          float g = 0.5f * v * (1.0f + erff(v * 0.70710678118654752f));
          outbf[(size_t)e * sOut + (size_t)m * ldOut + col] = f2bf(g);
        } else { // EP_PATCH: m = b*196 + p  ->  z row b*T + 1 + p, add pos[1+p]
          int b = m / NPATCH, pidx = m % NPATCH;
          int row = b * T_ + 1 + pidx;
          out[(size_t)e * sOut + (size_t)row * ldOut + col] =
              v + pos[(size_t)e * sPos + (size_t)(1 + pidx) * ldOut + col];
        }
      }
    }
  }
}

// LayerNorm over D=1024; one row per block; writes bf16 activations.
__global__ __launch_bounds__(256)
void ln_kernel(const float* __restrict__ z,
               const float* __restrict__ w, size_t sW,
               const float* __restrict__ b, size_t sB,
               unsigned short* __restrict__ h, int rowsPerE)
{
  __shared__ float red[256];
  const int tid = threadIdx.x;
  const int row = blockIdx.x;
  const int e   = row / rowsPerE;
  const float* x = z + (size_t)row * D_;
  float xv[4]; float s = 0.f;
  #pragma unroll
  for (int j = 0; j < 4; ++j) { xv[j] = x[tid + j * 256]; s += xv[j]; }
  red[tid] = s; __syncthreads();
  for (int o = 128; o > 0; o >>= 1) { if (tid < o) red[tid] += red[tid + o]; __syncthreads(); }
  float mean = red[0] * (1.0f / D_); __syncthreads();
  float vs = 0.f;
  #pragma unroll
  for (int j = 0; j < 4; ++j) { float d = xv[j] - mean; vs += d * d; }
  red[tid] = vs; __syncthreads();
  for (int o = 128; o > 0; o >>= 1) { if (tid < o) red[tid] += red[tid + o]; __syncthreads(); }
  float rstd = rsqrtf(red[0] * (1.0f / D_) + 1e-6f);
  const float* we = w + (size_t)e * sW;
  const float* be = b + (size_t)e * sB;
  #pragma unroll
  for (int j = 0; j < 4; ++j) {
    int i = tid + j * 256;
    h[(size_t)row * D_ + i] = f2bf((xv[j] - mean) * rstd * we[i] + be[i]);
  }
}

// softmax(q k^T / 8) per (e,b,h,t): grid (T, E*B*H)
__global__ __launch_bounds__(256)
void attn_scores(const float* __restrict__ qkv, float* __restrict__ S)
{
  __shared__ float q[DH_];
  __shared__ float red[256];
  const int tid = threadIdx.x;
  const int t   = blockIdx.x;
  const int ebh = blockIdx.y;
  const int eb  = ebh / H_, hh = ebh % H_;
  const float* base = qkv + (size_t)eb * T_ * (3 * D_);
  if (tid < DH_) q[tid] = base[(size_t)t * (3 * D_) + hh * DH_ + tid];
  __syncthreads();
  float val = -3.0e38f;
  if (tid < T_) {
    const float* kp = base + (size_t)tid * (3 * D_) + D_ + hh * DH_;
    float dot = 0.f;
    #pragma unroll 8
    for (int d = 0; d < DH_; ++d) dot += q[d] * kp[d];
    val = dot * 0.125f;   // 1/sqrt(64)
  }
  red[tid] = val; __syncthreads();
  for (int o = 128; o > 0; o >>= 1) { if (tid < o) red[tid] = fmaxf(red[tid], red[tid + o]); __syncthreads(); }
  float mx = red[0]; __syncthreads();
  float p = (tid < T_) ? __expf(val - mx) : 0.0f;
  red[tid] = p; __syncthreads();
  for (int o = 128; o > 0; o >>= 1) { if (tid < o) red[tid] += red[tid + o]; __syncthreads(); }
  float inv = 1.0f / red[0];
  if (tid < T_) S[((size_t)ebh * T_ + t) * T_ + tid] = p * inv;
}

// o[e,b,t,:] = att @ V  -> bf16 for the proj GEMM; grid ((T*D)/256, E*B)
__global__ __launch_bounds__(256)
void attn_out(const float* __restrict__ qkv, const float* __restrict__ S,
              unsigned short* __restrict__ o)
{
  const int eb  = blockIdx.y;
  const int idx = blockIdx.x * 256 + threadIdx.x;
  if (idx >= T_ * D_) return;
  const int t = idx >> 10, col = idx & (D_ - 1);
  const int hh = col >> 6, d = col & 63;
  const float* Sr = S + ((size_t)(eb * H_ + hh) * T_ + t) * T_;
  const float* vp = qkv + (size_t)eb * T_ * (3 * D_) + 2 * D_ + hh * DH_ + d;
  float acc = 0.f;
  for (int s = 0; s < T_; ++s) acc += Sr[s] * vp[(size_t)s * (3 * D_)];
  o[(size_t)(eb * T_ + t) * D_ + col] = f2bf(acc);
}

// x (B,3,224,224) -> patches bf16 [B, 196, 768]
__global__ __launch_bounds__(256)
void patch_extract(const float* __restrict__ x, unsigned short* __restrict__ pt)
{
  const int idx = blockIdx.x * 256 + threadIdx.x;
  const int total = B_ * NPATCH * PK_;
  if (idx >= total) return;
  int j = idx % PK_;
  int p = (idx / PK_) % NPATCH;
  int b = idx / (PK_ * NPATCH);
  int c = j >> 8, rem = j & 255;
  int py = rem >> 4, px = rem & 15;
  int gy = p / 14, gx = p % 14;
  float v = x[(((size_t)(b * 3 + c) * 224) + gy * 16 + py) * 224 + gx * 16 + px];
  pt[idx] = f2bf(v);
}

// z[e,b,0,:] = cls + pos[0]
__global__ __launch_bounds__(256)
void cls_pos(const float* __restrict__ cls, const float* __restrict__ pos,
             float* __restrict__ z)
{
  const int eb = blockIdx.x;
  const int e  = eb / B_;
  for (int i = threadIdx.x; i < D_; i += 256)
    z[(size_t)eb * T_ * D_ + i] = cls[(size_t)e * D_ + i] + pos[(size_t)e * T_ * D_ + i];
}

// final LN on token 0 + 4-class head; grid (E*B)
__global__ __launch_bounds__(256)
void final_head(const float* __restrict__ z, const float* __restrict__ lnw,
                const float* __restrict__ lnb, const float* __restrict__ hw,
                const float* __restrict__ hb, float* __restrict__ outp)
{
  __shared__ float red[256];
  __shared__ float hln[D_];
  const int tid = threadIdx.x;
  const int eb  = blockIdx.x;
  const int e   = eb / B_;
  const float* x = z + (size_t)eb * T_ * D_;    // token 0
  float xv[4]; float s = 0.f;
  #pragma unroll
  for (int j = 0; j < 4; ++j) { xv[j] = x[tid + j * 256]; s += xv[j]; }
  red[tid] = s; __syncthreads();
  for (int o = 128; o > 0; o >>= 1) { if (tid < o) red[tid] += red[tid + o]; __syncthreads(); }
  float mean = red[0] * (1.0f / D_); __syncthreads();
  float vs = 0.f;
  #pragma unroll
  for (int j = 0; j < 4; ++j) { float d = xv[j] - mean; vs += d * d; }
  red[tid] = vs; __syncthreads();
  for (int o = 128; o > 0; o >>= 1) { if (tid < o) red[tid] += red[tid + o]; __syncthreads(); }
  float rstd = rsqrtf(red[0] * (1.0f / D_) + 1e-6f); __syncthreads();
  #pragma unroll
  for (int j = 0; j < 4; ++j) {
    int i = tid + j * 256;
    hln[i] = (xv[j] - mean) * rstd * lnw[(size_t)e * D_ + i] + lnb[(size_t)e * D_ + i];
  }
  __syncthreads();
  for (int c = 0; c < C_; ++c) {
    float part = 0.f;
    const float* wr = hw + ((size_t)e * C_ + c) * D_;
    for (int i = tid; i < D_; i += 256) part += hln[i] * wr[i];
    red[tid] = part; __syncthreads();
    for (int o = 128; o > 0; o >>= 1) { if (tid < o) red[tid] += red[tid + o]; __syncthreads(); }
    if (tid == 0) outp[(size_t)eb * C_ + c] = red[0] + hb[(size_t)e * C_ + c];
    __syncthreads();
  }
}

extern "C" void kernel_launch(void* const* d_in, const int* in_sizes, int n_in,
                              void* d_out, int out_size, void* d_ws, size_t ws_size,
                              hipStream_t stream)
{
  (void)in_sizes; (void)n_in; (void)out_size; (void)ws_size;
  const float* x       = (const float*)d_in[0];
  const float* patch_w = (const float*)d_in[1];
  const float* patch_b = (const float*)d_in[2];
  const float* cls_tok = (const float*)d_in[3];
  const float* pos     = (const float*)d_in[4];
  const float* ln1_w   = (const float*)d_in[5];
  const float* ln1_b   = (const float*)d_in[6];
  const float* qkv_w   = (const float*)d_in[7];
  const float* qkv_b   = (const float*)d_in[8];
  const float* proj_w  = (const float*)d_in[9];
  const float* proj_b  = (const float*)d_in[10];
  const float* ls1     = (const float*)d_in[11];
  const float* ln2_w   = (const float*)d_in[12];
  const float* ln2_b   = (const float*)d_in[13];
  const float* fc1_w   = (const float*)d_in[14];
  const float* fc1_b   = (const float*)d_in[15];
  const float* fc2_w   = (const float*)d_in[16];
  const float* fc2_b   = (const float*)d_in[17];
  const float* ls2     = (const float*)d_in[18];
  const float* lnf_w   = (const float*)d_in[19];
  const float* lnf_b   = (const float*)d_in[20];
  const float* head_w  = (const float*)d_in[21];
  const float* head_b  = (const float*)d_in[22];

  // workspace carve (~34 MB)
  char* p = (char*)d_ws;
  auto carve = [&](size_t bytes) -> void* {
    void* r = (void*)p;
    p += (bytes + 255) & ~(size_t)255;
    return r;
  };
  float*          z     = (float*)carve((size_t)E_ * B_ * T_ * D_ * 4);          // residual (f32)
  float*          qkv   = (float*)carve((size_t)E_ * B_ * T_ * 3 * D_ * 4);      // qkv (f32)
  float*          S     = (float*)carve((size_t)E_ * B_ * H_ * T_ * T_ * 4);     // attn probs
  unsigned short* hbf   = (unsigned short*)carve((size_t)E_ * B_ * T_ * D_ * 2); // LN out (bf16)
  unsigned short* obf   = (unsigned short*)carve((size_t)E_ * B_ * T_ * D_ * 2); // attn out (bf16)
  unsigned short* mlpbf = (unsigned short*)carve((size_t)E_ * B_ * T_ * F_ * 2); // gelu(fc1) (bf16)
  unsigned short* ptbf  = (unsigned short*)carve((size_t)B_ * NPATCH * PK_ * 2); // patches (bf16)

  const int MT = B_ * T_;          // 394 rows per expert
  const int MB = (MT + 127) / 128; // 4 M-tiles
  const dim3 blk(256);

  patch_extract<<<dim3((B_ * NPATCH * PK_ + 255) / 256), blk, 0, stream>>>(x, ptbf);
  cls_pos<<<dim3(E_ * B_), blk, 0, stream>>>(cls_tok, pos, z);
  gemm_bf16_wmma<EP_PATCH><<<dim3(D_ / 64, (B_ * NPATCH + 127) / 128, E_), blk, 0, stream>>>(
      ptbf, 0, patch_w, (size_t)D_ * PK_, patch_b, (size_t)D_,
      z, (size_t)B_ * T_ * D_, D_, nullptr, nullptr, 0, pos, (size_t)T_ * D_,
      B_ * NPATCH, PK_);

  for (int l = 0; l < L_; ++l) {
    const float* l1w = ln1_w + (size_t)l * D_;
    const float* l1b = ln1_b + (size_t)l * D_;
    const float* qw  = qkv_w + (size_t)l * 3 * D_ * D_;
    const float* qb  = qkv_b + (size_t)l * 3 * D_;
    const float* pw  = proj_w + (size_t)l * D_ * D_;
    const float* pb  = proj_b + (size_t)l * D_;
    const float* s1  = ls1 + (size_t)l * D_;
    const float* l2w = ln2_w + (size_t)l * D_;
    const float* l2b = ln2_b + (size_t)l * D_;
    const float* f1w = fc1_w + (size_t)l * F_ * D_;
    const float* f1b = fc1_b + (size_t)l * F_;
    const float* f2w = fc2_w + (size_t)l * D_ * F_;
    const float* f2b = fc2_b + (size_t)l * D_;
    const float* s2  = ls2 + (size_t)l * D_;

    ln_kernel<<<dim3(E_ * B_ * T_), blk, 0, stream>>>(
        z, l1w, (size_t)L_ * D_, l1b, (size_t)L_ * D_, hbf, MT);
    gemm_bf16_wmma<EP_STORE><<<dim3(3 * D_ / 64, MB, E_), blk, 0, stream>>>(
        hbf, (size_t)MT * D_, qw, (size_t)L_ * 3 * D_ * D_, qb, (size_t)L_ * 3 * D_,
        qkv, (size_t)MT * 3 * D_, 3 * D_, nullptr, nullptr, 0, nullptr, 0, MT, D_);
    attn_scores<<<dim3(T_, E_ * B_ * H_), blk, 0, stream>>>(qkv, S);
    attn_out<<<dim3((T_ * D_ + 255) / 256, E_ * B_), blk, 0, stream>>>(qkv, S, obf);
    gemm_bf16_wmma<EP_RESID><<<dim3(D_ / 64, MB, E_), blk, 0, stream>>>(
        obf, (size_t)MT * D_, pw, (size_t)L_ * D_ * D_, pb, (size_t)L_ * D_,
        z, (size_t)MT * D_, D_, nullptr, s1, (size_t)L_ * D_, nullptr, 0, MT, D_);
    ln_kernel<<<dim3(E_ * B_ * T_), blk, 0, stream>>>(
        z, l2w, (size_t)L_ * D_, l2b, (size_t)L_ * D_, hbf, MT);
    gemm_bf16_wmma<EP_GELU><<<dim3(F_ / 64, MB, E_), blk, 0, stream>>>(
        hbf, (size_t)MT * D_, f1w, (size_t)L_ * F_ * D_, f1b, (size_t)L_ * F_,
        nullptr, (size_t)MT * F_, F_, mlpbf, nullptr, 0, nullptr, 0, MT, D_);
    gemm_bf16_wmma<EP_RESID><<<dim3(D_ / 64, MB, E_), blk, 0, stream>>>(
        mlpbf, (size_t)MT * F_, f2w, (size_t)L_ * D_ * F_, f2b, (size_t)L_ * D_,
        z, (size_t)MT * D_, D_, nullptr, s2, (size_t)L_ * D_, nullptr, 0, MT, F_);
  }

  final_head<<<dim3(E_ * B_), blk, 0, stream>>>(z, lnf_w, lnf_b, head_w, head_b,
                                                (float*)d_out);
}
